// Grapher_41489384079612
// MI455X (gfx1250) — compile-verified, matrix operand order checked by reading, not run on hardware
//
#include <hip/hip_runtime.h>
#include <hip/hip_bf16.h>

typedef __attribute__((ext_vector_type(16))) __bf16 v16bf;
typedef __attribute__((ext_vector_type(8)))  float  v8f;

#define BN_EPS 1e-5f

static __device__ __forceinline__ unsigned short f32_to_bf16_rne(float f) {
    unsigned int u = __float_as_uint(f);
    u += 0x7FFFu + ((u >> 16) & 1u);       // round-to-nearest-even
    return (unsigned short)(u >> 16);
}

// ---------------------------------------------------------------- utilities
__global__ void fill_u32_kernel(unsigned int* __restrict__ p, unsigned int v, size_t n) {
    size_t i = (size_t)blockIdx.x * blockDim.x + threadIdx.x;
    if (i < n) p[i] = v;
}

__global__ void f32_to_bf16_kernel(const float* __restrict__ in,
                                   unsigned short* __restrict__ out, size_t n) {
    size_t i = (size_t)blockIdx.x * blockDim.x + threadIdx.x;
    if (i < n) out[i] = f32_to_bf16_rne(in[i]);
}

// ---------------------------------------------------------------- WMMA GEMM
// C[N,M] = A[N,K](bf16) * W[M,K](bf16)^T + bias.  One wave -> 16 rows x 64 cols.
// K compile-time (256/512), M multiple of 64, N multiple of 16 (50000=3125*16).
// Register double-buffered k-loop: loads for step s+1 issue before WMMAs of
// step s so the compiler can wait with loadcnt>0 instead of draining.
static __device__ __forceinline__ void store_tile(float* __restrict__ C, v8f acc,
                                                  const float* __restrict__ bias,
                                                  int rbase, int col, int M) {
    const float bv = bias[col];
#pragma unroll
    for (int i = 0; i < 8; ++i)
        C[(size_t)(rbase + i) * M + col] = acc[i] + bv;
}

template <int K>
__global__ void __launch_bounds__(256)
gemm_bf16_wmma_kernel(const unsigned short* __restrict__ A,   // [N,K] bf16
                      const unsigned short* __restrict__ W,   // [M,K] bf16
                      const float* __restrict__ bias,         // [M]
                      float* __restrict__ C,                  // [N,M] f32
                      int N, int M)
{
    constexpr int NSTEP = K / 32;
    const int lane = threadIdx.x & 31;
    const int wave = threadIdx.x >> 5;
    const int rowTile = blockIdx.x * 8 + wave;        // 16-row tile index
    if (rowTile * 16 >= N) return;                    // wave-uniform exit
    const int r16  = lane & 15;
    const int hi   = lane >> 4;                       // half-lane selects K-half
    const int col0 = blockIdx.y * 64;
    const int row  = rowTile * 16 + r16;

    const unsigned int* Ar  = (const unsigned int*)(A + (size_t)row * K);
    const unsigned int* Wr0 = (const unsigned int*)(W + (size_t)(col0      + r16) * K);
    const unsigned int* Wr1 = (const unsigned int*)(W + (size_t)(col0 + 16 + r16) * K);
    const unsigned int* Wr2 = (const unsigned int*)(W + (size_t)(col0 + 32 + r16) * K);
    const unsigned int* Wr3 = (const unsigned int*)(W + (size_t)(col0 + 48 + r16) * K);

    union Frag { v16bf v; unsigned int u[8]; };
    Frag a[2], b0[2], b1[2], b2[2], b3[2];

    // 16-bit A/B fragment layout: lanes 0-15 hold K=[k0..k0+7] (VGPR0-3) and
    // K=[k0+16..k0+23] (VGPR4-7); lanes 16-31 hold K+8 / K+24.
    auto load_frags = [&](int s, int buf) {
        const int kd = s * 16 + hi * 4;               // dword (bf16-pair) index
#pragma unroll
        for (int i = 0; i < 4; ++i) {
            a[buf].u[i]      = Ar[kd + i];     a[buf].u[4 + i]  = Ar[kd + 8 + i];
            b0[buf].u[i]     = Wr0[kd + i];    b0[buf].u[4 + i] = Wr0[kd + 8 + i];
            b1[buf].u[i]     = Wr1[kd + i];    b1[buf].u[4 + i] = Wr1[kd + 8 + i];
            b2[buf].u[i]     = Wr2[kd + i];    b2[buf].u[4 + i] = Wr2[kd + 8 + i];
            b3[buf].u[i]     = Wr3[kd + i];    b3[buf].u[4 + i] = Wr3[kd + 8 + i];
        }
    };

    v8f acc0 = {}, acc1 = {}, acc2 = {}, acc3 = {};
    load_frags(0, 0);
#pragma unroll
    for (int s = 0; s < NSTEP; ++s) {
        const int cur = s & 1;
        const int nxt = cur ^ 1;
        if (s + 1 < NSTEP) load_frags(s + 1, nxt);    // prefetch next K-slice
        acc0 = __builtin_amdgcn_wmma_f32_16x16x32_bf16(false, a[cur].v, false, b0[cur].v, (short)0, acc0, false, false);
        acc1 = __builtin_amdgcn_wmma_f32_16x16x32_bf16(false, a[cur].v, false, b1[cur].v, (short)0, acc1, false, false);
        acc2 = __builtin_amdgcn_wmma_f32_16x16x32_bf16(false, a[cur].v, false, b2[cur].v, (short)0, acc2, false, false);
        acc3 = __builtin_amdgcn_wmma_f32_16x16x32_bf16(false, a[cur].v, false, b3[cur].v, (short)0, acc3, false, false);
    }

    // D layout: lane column = r16; rows = hi*8 + i for VGPR i
    const int rbase = rowTile * 16 + hi * 8;
    store_tile(C, acc0, bias, rbase, col0 +  0 + r16, M);
    store_tile(C, acc1, bias, rbase, col0 + 16 + r16, M);
    store_tile(C, acc2, bias, rbase, col0 + 32 + r16, M);
    store_tile(C, acc3, bias, rbase, col0 + 48 + r16, M);
}

// ---------------------------------------------------------------- BN stats
__global__ void col_stats_kernel(const float* __restrict__ X, int N, int C,
                                 float* __restrict__ sums, float* __restrict__ sumsqs,
                                 int rowsPerBlock)
{
    int c = blockIdx.x * blockDim.x + threadIdx.x;
    if (c >= C) return;
    int r0 = blockIdx.y * rowsPerBlock;
    int r1 = min(N, r0 + rowsPerBlock);
    float s = 0.f, q = 0.f;
    for (int r = r0; r < r1; ++r) {
        float v = X[(size_t)r * C + c];
        s += v; q += v * v;
    }
    atomicAdd(&sums[c], s);
    atomicAdd(&sumsqs[c], q);
}

// ------------------------------------------------- BN apply (+GELU/residual)
__global__ void bn_apply_kernel(const float* __restrict__ X, int N, int C,
                                const float* __restrict__ sums, const float* __restrict__ sumsqs,
                                const float* __restrict__ gamma, const float* __restrict__ beta,
                                const float* __restrict__ residual,   // may be null
                                int do_gelu,
                                float* __restrict__ outF,             // may be null
                                unsigned short* __restrict__ outBf,   // may be null
                                int outBfStride)
{
    size_t idx = (size_t)blockIdx.x * blockDim.x + threadIdx.x;
    size_t total = (size_t)N * C;
    if (idx >= total) return;
    int c = (int)(idx % C);
    size_t row = idx / C;
    float invN = 1.0f / (float)N;
    float mu  = sums[c] * invN;
    float var = sumsqs[c] * invN - mu * mu;     // biased variance (torch-style)
    float y = (X[idx] - mu) * rsqrtf(var + BN_EPS) * gamma[c] + beta[c];
    if (do_gelu) {   // tanh-approx gelu (jax.nn.gelu default)
        float t = 0.7978845608028654f * (y + 0.044715f * y * y * y);
        y = 0.5f * y * (1.0f + tanhf(t));
    }
    if (residual) y += residual[idx];
    if (outF)  outF[idx] = y;
    if (outBf) outBf[row * (size_t)outBfStride + c] = f32_to_bf16_rne(y);
}

// ---------------------------------------------------------------- edge max
// segment_max(h[src]-h[dst], dst) via atomicMax on monotone uint float encoding
__global__ void edge_max_kernel(const float* __restrict__ h,
                                const int* __restrict__ src, const int* __restrict__ dst,
                                unsigned int* __restrict__ menc, int E, int C)
{
    int c = threadIdx.x;          // blockDim.x == C
    for (int e = blockIdx.x; e < E; e += gridDim.x) {
        int s = src[e], d = dst[e];
        float diff = h[(size_t)s * C + c] - h[(size_t)d * C + c];
        unsigned int u = __float_as_uint(diff);
        u = (u & 0x80000000u) ? ~u : (u | 0x80000000u);
        atomicMax(&menc[(size_t)d * C + c], u);
    }
}

// -------------------------------------- hg = concat(h, decode(m)) as bf16
__global__ void assemble_hg_kernel(const float* __restrict__ h,
                                   const unsigned int* __restrict__ menc,
                                   unsigned short* __restrict__ hg, int N, int C)
{
    size_t idx = (size_t)blockIdx.x * blockDim.x + threadIdx.x;
    size_t total = (size_t)N * (2 * C);
    if (idx >= total) return;
    int c = (int)(idx % (2 * C));
    size_t row = idx / (2 * C);
    float v;
    if (c < C) {
        v = h[row * C + c];
    } else {
        unsigned int u = menc[row * C + (c - C)];
        unsigned int bits = (u & 0x80000000u) ? (u & 0x7FFFFFFFu) : ~u;
        if (((bits >> 23) & 0xFFu) == 0xFFu) bits = 0u;   // -inf/NaN (no in-edges) -> 0
        v = __uint_as_float(bits);
    }
    hg[idx] = f32_to_bf16_rne(v);
}

// ---------------------------------------------------------------- launcher
extern "C" void kernel_launch(void* const* d_in, const int* in_sizes, int n_in,
                              void* d_out, int out_size, void* d_ws, size_t ws_size,
                              hipStream_t stream)
{
    (void)n_in; (void)out_size; (void)ws_size;
    const float* x   = (const float*)d_in[0];
    const int*  eidx = (const int*)d_in[1];
    const float* w1  = (const float*)d_in[2];
    const float* b1  = (const float*)d_in[3];
    const float* g1  = (const float*)d_in[4];
    const float* bt1 = (const float*)d_in[5];
    const float* wg  = (const float*)d_in[6];
    const float* bg  = (const float*)d_in[7];
    const float* gg  = (const float*)d_in[8];
    const float* btg = (const float*)d_in[9];
    const float* w2  = (const float*)d_in[10];
    const float* b2  = (const float*)d_in[11];
    const float* g2  = (const float*)d_in[12];
    const float* bt2 = (const float*)d_in[13];
    float* out = (float*)d_out;

    const int Cc = 256;
    const int N  = in_sizes[0] / Cc;     // 50000
    const int E  = in_sizes[1] / 2;      // 800000
    const int CG = 512;

    // bump allocator over workspace
    char* p = (char*)d_ws;
    auto alloc = [&](size_t bytes) -> void* {
        void* q = (void*)p;
        p += (bytes + 255) & ~(size_t)255;
        return q;
    };
    unsigned short* xbf  = (unsigned short*)alloc((size_t)N * Cc * 2);
    unsigned short* w1bf = (unsigned short*)alloc((size_t)256 * Cc * 2);
    unsigned short* wgbf = (unsigned short*)alloc((size_t)CG * CG * 2);
    unsigned short* w2bf = (unsigned short*)alloc((size_t)Cc * CG * 2);
    float*        h      = (float*)alloc((size_t)N * Cc * 4);   // h_pre -> h_bn -> fc2 pre
    unsigned int* menc   = (unsigned int*)alloc((size_t)N * Cc * 4);
    unsigned short* hgbf = (unsigned short*)alloc((size_t)N * CG * 2);
    float*        hgpre  = (float*)alloc((size_t)N * CG * 4);
    float*        stats  = (float*)alloc((size_t)2 * CG * 4);
    float* sums  = stats;
    float* sumsq = stats + CG;

    const int T = 256;
    auto nblk = [&](size_t n) { return (unsigned int)((n + T - 1) / T); };
    const int rowTiles = (N + 15) / 16;            // 3125
    const int ROWS_PER_BLOCK = 512;
    const unsigned int statRowBlocks = (unsigned int)((N + ROWS_PER_BLOCK - 1) / ROWS_PER_BLOCK);

    // --- bf16 conversions of x and all weights
    f32_to_bf16_kernel<<<nblk((size_t)N * Cc), T, 0, stream>>>(x,  xbf,  (size_t)N * Cc);
    f32_to_bf16_kernel<<<nblk((size_t)256 * Cc), T, 0, stream>>>(w1, w1bf, (size_t)256 * Cc);
    f32_to_bf16_kernel<<<nblk((size_t)CG * CG),  T, 0, stream>>>(wg, wgbf, (size_t)CG * CG);
    f32_to_bf16_kernel<<<nblk((size_t)Cc * CG),  T, 0, stream>>>(w2, w2bf, (size_t)Cc * CG);

    // --- fc1: h = x @ w1^T + b1   (K=256)
    {
        dim3 grid((rowTiles + 7) / 8, 256 / 64);
        gemm_bf16_wmma_kernel<256><<<grid, 256, 0, stream>>>(xbf, w1bf, b1, h, N, 256);
    }
    // --- BN1 (in place)
    fill_u32_kernel<<<nblk(2 * CG), T, 0, stream>>>((unsigned int*)stats, 0u, (size_t)2 * CG);
    {
        dim3 gs(nblk(Cc), statRowBlocks);
        col_stats_kernel<<<gs, T, 0, stream>>>(h, N, Cc, sums, sumsq, ROWS_PER_BLOCK);
    }
    bn_apply_kernel<<<nblk((size_t)N * Cc), T, 0, stream>>>(
        h, N, Cc, sums, sumsq, g1, bt1, nullptr, 0, h, nullptr, 0);

    // --- graph max-relative conv
    fill_u32_kernel<<<nblk((size_t)N * Cc), T, 0, stream>>>(menc, 0x007FFFFFu /*enc(-inf)*/,
                                                            (size_t)N * Cc);
    edge_max_kernel<<<E, Cc, 0, stream>>>(h, eidx, eidx + E, menc, E, Cc);
    assemble_hg_kernel<<<nblk((size_t)N * CG), T, 0, stream>>>(h, menc, hgbf, N, Cc);

    // --- graph fc: hg_pre = hg @ wg^T + bg   (K=512)
    {
        dim3 grid((rowTiles + 7) / 8, CG / 64);
        gemm_bf16_wmma_kernel<512><<<grid, 256, 0, stream>>>(hgbf, wgbf, bg, hgpre, N, CG);
    }
    // --- BN + GELU, emit bf16 for fc2 (reuse hgbf)
    fill_u32_kernel<<<nblk(2 * CG), T, 0, stream>>>((unsigned int*)stats, 0u, (size_t)2 * CG);
    {
        dim3 gs(nblk(CG), statRowBlocks);
        col_stats_kernel<<<gs, T, 0, stream>>>(hgpre, N, CG, sums, sumsq, ROWS_PER_BLOCK);
    }
    bn_apply_kernel<<<nblk((size_t)N * CG), T, 0, stream>>>(
        hgpre, N, CG, sums, sumsq, gg, btg, nullptr, 1, nullptr, hgbf, CG);

    // --- fc2: pre = hg @ w2^T + b2 (reuse h buffer)  (K=512)
    {
        dim3 grid((rowTiles + 7) / 8, 256 / 64);
        gemm_bf16_wmma_kernel<512><<<grid, 256, 0, stream>>>(hgbf, w2bf, b2, h, N, 256);
    }
    // --- BN2 + residual -> out
    fill_u32_kernel<<<nblk(2 * CG), T, 0, stream>>>((unsigned int*)stats, 0u, (size_t)2 * CG);
    {
        dim3 gs(nblk(Cc), statRowBlocks);
        col_stats_kernel<<<gs, T, 0, stream>>>(h, N, Cc, sums, sumsq, ROWS_PER_BLOCK);
    }
    bn_apply_kernel<<<nblk((size_t)N * Cc), T, 0, stream>>>(
        h, N, Cc, sums, sumsq, g2, bt2, x, 0, out, nullptr, 0);
}
